// ViT_27144193311303
// MI455X (gfx1250) — compile-verified, matrix operand order checked by reading, not run on hardware
//
#include <hip/hip_runtime.h>

#define E    256
#define FF   2048
#define LYR  6
#define NH   8
#define TT   64
#define MP   80      // padded row count for M=65 decode buffers (multiple of 16)
#define INP  224
#define KE   26
#define SE   13
#define KD   29
#define SD   13

typedef __attribute__((ext_vector_type(16))) _Float16 v16h;
typedef __attribute__((ext_vector_type(8)))  float    v8f;

// ---------------------------------------------------------------------------
// WMMA GEMM: C = act(A[MpadxK] @ W[NxK]^T + bias[N]) (+ Res[MpadxN])
// All buffers are padded to a multiple of 16 rows -> zero divergence:
// loads/stores fully unconditional, EXEC stays all-1s (WMMA requirement).
//
// Fragment load trick: for lane group g = lane>>4, the 16-bit A(16x32)
// fragment's K indices are two contiguous runs of 8 floats:
//   elements 0..7  -> K = k0 + 8g + e
//   elements 8..15 -> K = k0 + 16 + 8g + e
// Same map for the B fragment => 2x 32-byte vector loads per fragment.
// ---------------------------------------------------------------------------
template<int ACT, bool HAS_RES>   // ACT: 0=none, 1=relu, 2=tanh
__global__ void gemm_wmma_kernel(const float* __restrict__ A,
                                 const float* __restrict__ W,
                                 const float* __restrict__ bias,
                                 const float* __restrict__ Res,
                                 float* __restrict__ C,
                                 int N, int K)
{
  const int lane = threadIdx.x & 31;
  const int wave = threadIdx.x >> 5;
  const int tn   = blockIdx.x * 4 + wave;   // N/16 tiles, 4 per block
  const int tm   = blockIdx.y;
  const int g    = lane >> 4;
  const int l15  = lane & 15;

  const float* Arow = A + (size_t)(tm * 16 + l15) * K + 8 * g;
  const float* Wrow = W + (size_t)(tn * 16 + l15) * K + 8 * g;

  v8f acc = {};
  for (int k0 = 0; k0 < K; k0 += 32) {
    if (k0 + 32 < K) {
      // locality 3 -> WGP-scope prefetch (pulls into all cache levels)
      __builtin_prefetch(Wrow + k0 + 32, 0, 3);
      __builtin_prefetch(Arow + k0 + 32, 0, 3);
    }
    const v8f alo = *(const v8f*)(Arow + k0);
    const v8f ahi = *(const v8f*)(Arow + k0 + 16);
    const v8f blo = *(const v8f*)(Wrow + k0);
    const v8f bhi = *(const v8f*)(Wrow + k0 + 16);
    v16h a, b;
#pragma unroll
    for (int e = 0; e < 8; ++e) {
      a[e]     = (_Float16)alo[e];
      a[e + 8] = (_Float16)ahi[e];
      b[e]     = (_Float16)blo[e];
      b[e + 8] = (_Float16)bhi[e];
    }
    acc = __builtin_amdgcn_wmma_f32_16x16x32_f16(
        /*neg_a=*/false, a, /*neg_b=*/false, b,
        /*c_mod=*/(short)0, acc, /*reuse_a=*/false, /*reuse_b=*/false);
  }

  const int c = tn * 16 + l15;
  const float bv = bias[c];
#pragma unroll
  for (int v = 0; v < 8; ++v) {             // C layout: VGPR v -> M = v + 8*g
    const int r = tm * 16 + v + 8 * g;
    float val = acc[v] + bv;
    if (HAS_RES) val += Res[(size_t)r * N + c];
    if (ACT == 1) val = fmaxf(val, 0.0f);
    if (ACT == 2) val = tanhf(val);
    C[(size_t)r * N + c] = val;
  }
}

// ---------------------------------------------------------------------------
// Multi-head attention, one wave per (query t, head h). Dh = 32 = wave width.
// K rows are read as 4x 32-byte vectors (all bases/strides are 32B-aligned).
// Scores j > maskLimit get -1e9 (matches reference mask semantics).
// ---------------------------------------------------------------------------
__global__ void attn_kernel(const float* __restrict__ Q,  int qs,
                            const float* __restrict__ Kb, int ks,
                            const float* __restrict__ Vb, int vs,
                            float* __restrict__ O,        int os,
                            int Mk, int maskLimit)
{
  const int t = blockIdx.x, h = blockIdx.y, lane = threadIdx.x;
  __shared__ float sQ[32];
  __shared__ float sS[80];

  sQ[lane] = Q[(size_t)t * qs + h * 32 + lane];
  __syncthreads();

  for (int j = lane; j < Mk; j += 32) {
    const v8f* kr = (const v8f*)(Kb + (size_t)j * ks + h * 32);
    float s = 0.0f;
#pragma unroll
    for (int c4 = 0; c4 < 4; ++c4) {
      const v8f kk = kr[c4];
#pragma unroll
      for (int e = 0; e < 8; ++e) s += sQ[c4 * 8 + e] * kk[e];
    }
    s *= 0.17677669529663687f;              // 1/sqrt(Dh=32)
    if (j > maskLimit) s = -1.0e9f;
    sS[j] = s;
  }
  __syncthreads();

  float m = -3.0e38f;
  for (int j = lane; j < Mk; j += 32) m = fmaxf(m, sS[j]);
  for (int off = 16; off > 0; off >>= 1) m = fmaxf(m, __shfl_xor(m, off, 32));

  float sum = 0.0f;
  for (int j = lane; j < Mk; j += 32) sum += __expf(sS[j] - m);
  for (int off = 16; off > 0; off >>= 1) sum += __shfl_xor(sum, off, 32);
  const float inv = 1.0f / sum;

  for (int j = lane; j < Mk; j += 32) sS[j] = __expf(sS[j] - m) * inv;
  __syncthreads();

  float o = 0.0f;
  for (int j = 0; j < Mk; ++j) o += sS[j] * Vb[(size_t)j * vs + h * 32 + lane];
  O[(size_t)t * os + h * 32 + lane] = o;
}

// Row-wise LayerNorm over E=256, one block per row.
__global__ void layernorm_kernel(const float* __restrict__ X,
                                 const float* __restrict__ g,
                                 const float* __restrict__ b,
                                 float* __restrict__ Y)
{
  const int r = blockIdx.x, t = threadIdx.x;
  __shared__ float red[E];
  const float v = X[(size_t)r * E + t];
  red[t] = v;  __syncthreads();
  for (int s = E / 2; s > 0; s >>= 1) { if (t < s) red[t] += red[t + s]; __syncthreads(); }
  const float mean = red[0] / (float)E;
  __syncthreads();
  const float d = v - mean;
  red[t] = d * d;  __syncthreads();
  for (int s = E / 2; s > 0; s >>= 1) { if (t < s) red[t] += red[t + s]; __syncthreads(); }
  const float var = red[0] / (float)E;
  Y[(size_t)r * E + t] = d * rsqrtf(var + 1e-5f) * g[t] + b[t];
}

// Conv encoder: 64 channels x 16x16 out, tanh(conv + b) -> h_flat[64][256]
__global__ void conv_enc_kernel(const float* __restrict__ x,
                                const float* __restrict__ w,
                                const float* __restrict__ bias,
                                float* __restrict__ h)
{
  const int oc = blockIdx.x;
  const int p  = threadIdx.x;               // oy*16 + ox
  const int oy = p >> 4, ox = p & 15;
  float s = bias[oc];
  const int iy0 = oy * SE, ix0 = ox * SE;
  for (int c = 0; c < 3; ++c)
    for (int ky = 0; ky < KE; ++ky) {
      const float* xr = x + ((size_t)c * INP + iy0 + ky) * INP + ix0;
      const float* wr = w + (((size_t)oc * 3 + c) * KE + ky) * KE;
      for (int kx = 0; kx < KE; ++kx) s += xr[kx] * wr[kx];
    }
  h[(size_t)oc * 256 + p] = tanhf(s);
}

// Transposed conv: stride 13, kernel 29, out 3x224x224.
__global__ void convt_kernel(const float* __restrict__ h2,
                             const float* __restrict__ w,
                             const float* __restrict__ bias,
                             float* __restrict__ y)
{
  const int idx = blockIdx.x * blockDim.x + threadIdx.x;
  if (idx >= 3 * INP * INP) return;
  const int c = idx / (INP * INP);
  const int r = idx % (INP * INP);
  const int Y = r / INP, X = r % INP;
  float s = bias[c];
  for (int yy = 0; yy < 16; ++yy) {
    const int ky = Y - SD * yy;
    if (ky < 0 || ky >= KD) continue;
    for (int xx = 0; xx < 16; ++xx) {
      const int kx = X - SD * xx;
      if (kx < 0 || kx >= KD) continue;
      const float* wp = w + ((size_t)c * KD + ky) * KD + kx;   // + t*3*29*29
      const float* hp = h2 + yy * 16 + xx;                      // + t*256
      for (int t = 0; t < TT; ++t)
        s += hp[(size_t)t * 256] * wp[(size_t)t * 3 * KD * KD];
    }
  }
  y[idx] = s;
}

__global__ void zero_kernel(float* p, int n) {
  const int i = blockIdx.x * 256 + threadIdx.x;
  if (i < n) p[i] = 0.0f;
}

__global__ void copy_row_kernel(const float* __restrict__ src, float* __restrict__ dst) {
  dst[threadIdx.x] = src[threadIdx.x];
}

// ---------------------------------------------------------------------------
// Host side
// ---------------------------------------------------------------------------
struct EncP {
  const float *qkv_w,*qkv_b,*out_w,*out_b,*ff1_w,*ff1_b,*ff2_w,*ff2_b;
  const float *ln1_g,*ln1_b,*ln2_g,*ln2_b,*fn_g,*fn_b;
};
struct DecP {
  const float *sa_qkv_w,*sa_qkv_b,*sa_out_w,*sa_out_b;
  const float *ca_qkv_w,*ca_qkv_b,*ca_out_w,*ca_out_b;
  const float *ff1_w,*ff1_b,*ff2_w,*ff2_b;
  const float *ln1_g,*ln1_b,*ln2_g,*ln2_b,*ln3_g,*ln3_b,*fn_g,*fn_b;
};
struct Ws {
  float *h_flat,*seq,*mem_p,*mem_q,*buf,*x,*t_qkv,*t_q,*t_att,*t_res,*t_ff,*t_kvc,*dec_out,*h2;
};

static void launch_gemm(const float* A, const float* W, const float* bias, const float* Res,
                        float* C, int M, int N, int K, int act, hipStream_t st)
{
  dim3 grid(N / 64, (M + 15) / 16), block(128);   // pad rows are computed, never read back
  if (Res)            gemm_wmma_kernel<0, true ><<<grid, block, 0, st>>>(A, W, bias, Res,     C, N, K);
  else if (act == 1)  gemm_wmma_kernel<1, false><<<grid, block, 0, st>>>(A, W, bias, nullptr, C, N, K);
  else if (act == 2)  gemm_wmma_kernel<2, false><<<grid, block, 0, st>>>(A, W, bias, nullptr, C, N, K);
  else                gemm_wmma_kernel<0, false><<<grid, block, 0, st>>>(A, W, bias, nullptr, C, N, K);
}

static void run_encoder(const float* seq, const EncP& p, float* mem, const Ws& w, hipStream_t st)
{
  hipMemcpyAsync(w.x, seq, (size_t)TT * E * sizeof(float), hipMemcpyDeviceToDevice, st);
  for (int l = 0; l < LYR; ++l) {
    launch_gemm(w.x, p.qkv_w + (size_t)l*3*E*E, p.qkv_b + (size_t)l*3*E, nullptr, w.t_qkv, TT, 3*E, E, 0, st);
    attn_kernel<<<dim3(TT, NH), 32, 0, st>>>(w.t_qkv, 3*E, w.t_qkv + E, 3*E, w.t_qkv + 2*E, 3*E,
                                             w.t_att, E, TT, TT - 1);
    launch_gemm(w.t_att, p.out_w + (size_t)l*E*E, p.out_b + (size_t)l*E, w.x, w.t_res, TT, E, E, 0, st);
    layernorm_kernel<<<TT, E, 0, st>>>(w.t_res, p.ln1_g + l*E, p.ln1_b + l*E, w.x);
    launch_gemm(w.x, p.ff1_w + (size_t)l*FF*E, p.ff1_b + (size_t)l*FF, nullptr, w.t_ff, TT, FF, E, 1, st);
    launch_gemm(w.t_ff, p.ff2_w + (size_t)l*E*FF, p.ff2_b + (size_t)l*E, w.x, w.t_res, TT, E, FF, 0, st);
    layernorm_kernel<<<TT, E, 0, st>>>(w.t_res, p.ln2_g + l*E, p.ln2_b + l*E, w.x);
  }
  layernorm_kernel<<<TT, E, 0, st>>>(w.x, p.fn_g, p.fn_b, mem);
}

// Cross-attention K/V depend only on mem + layer weights: compute once per context.
static void build_cross_kv(const float* mem, const DecP& p, float* kvc, hipStream_t st)
{
  for (int l = 0; l < LYR; ++l)
    launch_gemm(mem, p.ca_qkv_w + (size_t)l*3*E*E + (size_t)E*E, p.ca_qkv_b + (size_t)l*3*E + E,
                nullptr, kvc + (size_t)l*TT*2*E, TT, 2*E, E, 0, st);
}

static void run_decoder(const float* tgt, int M, const DecP& p, const float* kvc,
                        int maskLimit, float* out, const Ws& w, hipStream_t st)
{
  hipMemcpyAsync(w.x, tgt, (size_t)M * E * sizeof(float), hipMemcpyDeviceToDevice, st);
  for (int l = 0; l < LYR; ++l) {
    // self attention (causal mask to maskLimit)
    launch_gemm(w.x, p.sa_qkv_w + (size_t)l*3*E*E, p.sa_qkv_b + (size_t)l*3*E, nullptr, w.t_qkv, M, 3*E, E, 0, st);
    attn_kernel<<<dim3(M, NH), 32, 0, st>>>(w.t_qkv, 3*E, w.t_qkv + E, 3*E, w.t_qkv + 2*E, 3*E,
                                            w.t_att, E, M, maskLimit);
    launch_gemm(w.t_att, p.sa_out_w + (size_t)l*E*E, p.sa_out_b + (size_t)l*E, w.x, w.t_res, M, E, E, 0, st);
    layernorm_kernel<<<M, E, 0, st>>>(w.t_res, p.ln1_g + l*E, p.ln1_b + l*E, w.x);
    // cross attention (keys/values precomputed from mem)
    launch_gemm(w.x, p.ca_qkv_w + (size_t)l*3*E*E, p.ca_qkv_b + (size_t)l*3*E, nullptr, w.t_q, M, E, E, 0, st);
    const float* kv = kvc + (size_t)l*TT*2*E;
    attn_kernel<<<dim3(M, NH), 32, 0, st>>>(w.t_q, E, kv, 2*E, kv + E, 2*E,
                                            w.t_att, E, TT, TT - 1);
    launch_gemm(w.t_att, p.ca_out_w + (size_t)l*E*E, p.ca_out_b + (size_t)l*E, w.x, w.t_res, M, E, E, 0, st);
    layernorm_kernel<<<M, E, 0, st>>>(w.t_res, p.ln2_g + l*E, p.ln2_b + l*E, w.x);
    // FFN
    launch_gemm(w.x, p.ff1_w + (size_t)l*FF*E, p.ff1_b + (size_t)l*FF, nullptr, w.t_ff, M, FF, E, 1, st);
    launch_gemm(w.t_ff, p.ff2_w + (size_t)l*E*FF, p.ff2_b + (size_t)l*E, w.x, w.t_res, M, E, FF, 0, st);
    layernorm_kernel<<<M, E, 0, st>>>(w.t_res, p.ln3_g + l*E, p.ln3_b + l*E, w.x);
  }
  layernorm_kernel<<<M, E, 0, st>>>(w.x, p.fn_g, p.fn_b, out);
}

extern "C" void kernel_launch(void* const* d_in, const int* in_sizes, int n_in,
                              void* d_out, int out_size, void* d_ws, size_t ws_size,
                              hipStream_t stream)
{
  (void)in_sizes; (void)n_in; (void)out_size; (void)ws_size;
  int idx = 0;
  auto nxt = [&]() { return (const float*)d_in[idx++]; };

  const float* xin    = nxt();
  const float* conv_w = nxt();
  const float* conv_b = nxt();
  const float* lin_w  = nxt();
  const float* lin_b  = nxt();

  auto load_enc = [&](EncP& p) {
    p.qkv_w = nxt(); p.qkv_b = nxt(); p.out_w = nxt(); p.out_b = nxt();
    p.ff1_w = nxt(); p.ff1_b = nxt(); p.ff2_w = nxt(); p.ff2_b = nxt();
    p.ln1_g = nxt(); p.ln1_b = nxt(); p.ln2_g = nxt(); p.ln2_b = nxt();
    p.fn_g  = nxt(); p.fn_b  = nxt();
  };
  auto load_dec = [&](DecP& p) {
    p.sa_qkv_w = nxt(); p.sa_qkv_b = nxt(); p.sa_out_w = nxt(); p.sa_out_b = nxt();
    p.ca_qkv_w = nxt(); p.ca_qkv_b = nxt(); p.ca_out_w = nxt(); p.ca_out_b = nxt();
    p.ff1_w = nxt(); p.ff1_b = nxt(); p.ff2_w = nxt(); p.ff2_b = nxt();
    p.ln1_g = nxt(); p.ln1_b = nxt(); p.ln2_g = nxt(); p.ln2_b = nxt();
    p.ln3_g = nxt(); p.ln3_b = nxt(); p.fn_g = nxt(); p.fn_b = nxt();
  };

  EncP prior_enc;  load_enc(prior_enc);
  DecP prior_dec;  load_dec(prior_dec);
  EncP post_enc;   load_enc(post_enc);
  DecP post_dec;   load_dec(post_dec);

  const float* dlin_w  = nxt();
  const float* dlin_b  = nxt();
  const float* convT_w = nxt();
  const float* convT_b = nxt();

  // Workspace bump allocation (floats). All GEMM operand buffers are padded
  // to MP=80 rows so the WMMA kernel never needs divergent guards.
  float* base = (float*)d_ws;
  size_t off = 0;
  auto alloc = [&](size_t n) { float* p = base + off; off += n; return p; };
  Ws w;
  w.h_flat  = alloc((size_t)TT * E);
  w.seq     = alloc((size_t)TT * E);
  w.mem_p   = alloc((size_t)TT * E);
  w.mem_q   = alloc((size_t)TT * E);
  w.buf     = alloc((size_t)MP * E);
  w.x       = alloc((size_t)MP * E);
  w.t_qkv   = alloc((size_t)MP * 3 * E);
  w.t_q     = alloc((size_t)MP * E);
  w.t_att   = alloc((size_t)MP * E);
  w.t_res   = alloc((size_t)MP * E);
  w.t_ff    = alloc((size_t)MP * FF);
  w.t_kvc   = alloc((size_t)LYR * TT * 2 * E);
  w.dec_out = alloc((size_t)MP * E);
  w.h2      = alloc((size_t)TT * E);

  // 1) conv patch encoder + tanh
  conv_enc_kernel<<<TT, 256, 0, stream>>>(xin, conv_w, conv_b, w.h_flat);
  // 2) seq = h_flat @ lin_w^T + lin_b
  launch_gemm(w.h_flat, lin_w, lin_b, nullptr, w.seq, TT, E, 256, 0, stream);
  // 3) prior encode
  run_encoder(w.seq, prior_enc, w.mem_p, w, stream);
  // 4) cross K/V cache for prior decoder
  build_cross_kv(w.mem_p, prior_dec, w.t_kvc, stream);
  // 5) autoregressive generation: buf[i+1] = decode(buf, mask<=i)[i]
  {
    const int n = MP * E;
    zero_kernel<<<(n + 255) / 256, 256, 0, stream>>>(w.buf, n);
  }
  for (int i = 0; i < TT; ++i) {
    run_decoder(w.buf, TT + 1, prior_dec, w.t_kvc, i, w.dec_out, w, stream);
    copy_row_kernel<<<1, E, 0, stream>>>(w.dec_out + (size_t)i * E, w.buf + (size_t)(i + 1) * E);
  }
  // 6) posterior encode + decode of gen = buf[1:]
  run_encoder(w.seq, post_enc, w.mem_q, w, stream);
  build_cross_kv(w.mem_q, post_dec, w.t_kvc, stream);
  run_decoder(w.buf + E, TT, post_dec, w.t_kvc, TT - 1, w.dec_out, w, stream);
  // 7) h2 = tanh(out @ dlin_w^T + dlin_b)
  launch_gemm(w.dec_out, dlin_w, dlin_b, nullptr, w.h2, TT, 256, E, 2, stream);
  // 8) transposed conv -> output 3x224x224
  {
    const int n = 3 * INP * INP;
    convt_kernel<<<(n + 255) / 256, 256, 0, stream>>>(w.h2, convT_w, convT_b, (float*)d_out);
  }
}
